// BPTTRTU_41274635714701
// MI455X (gfx1250) — compile-verified, matrix operand order checked by reading, not run on hardware
//
#include <hip/hip_runtime.h>

typedef unsigned short ushort_t;
typedef unsigned int   uint_t;

typedef __attribute__((ext_vector_type(16))) __bf16 v16bf;
typedef __attribute__((ext_vector_type(8)))  __bf16 v8bf;
typedef __attribute__((ext_vector_type(8)))  float  v8f;

#define T_DIM   8192
#define H_DIM   2048
#define D_DIM   1024
#define CHUNKS  64
#define CLEN    128   /* T_DIM / CHUNKS, = 2^7 */

__device__ __forceinline__ ushort_t f2bf(float f) {
    uint_t u = __float_as_uint(f);
    uint_t r = u + 0x7FFFu + ((u >> 16) & 1u);   // round-to-nearest-even
    return (ushort_t)(r >> 16);
}

// ---------------------------------------------------------------------------
// Per-channel parameters
// ---------------------------------------------------------------------------
__global__ void param_kernel(const float* __restrict__ lam,
                             const float* __restrict__ th,
                             float* __restrict__ aA, float* __restrict__ bA,
                             float* __restrict__ gA) {
    int h = blockIdx.x * blockDim.x + threadIdx.x;
    float y = expf(-expf(lam[h]));
    float z = expf(th[h]);
    float g = sqrtf(fmaxf(0.0f, 1.0f - y * y));
    aA[h] = y * cosf(z);
    bA[h] = y * sinf(z);
    gA[h] = g;
}

// ---------------------------------------------------------------------------
// f32 -> bf16 conversion (X), and gamma-scaled conversion (B matrices)
// ---------------------------------------------------------------------------
__global__ void cvt_x_kernel(const float* __restrict__ X, ushort_t* __restrict__ Xb) {
    int i = blockIdx.x * blockDim.x + threadIdx.x;        // group of 4 floats
    float4 f = ((const float4*)X)[i];
    uint_t lo = (uint_t)f2bf(f.x) | ((uint_t)f2bf(f.y) << 16);
    uint_t hi = (uint_t)f2bf(f.z) | ((uint_t)f2bf(f.w) << 16);
    ((uint2*)Xb)[i] = make_uint2(lo, hi);
}

__global__ void cvt_b_kernel(const float* __restrict__ B, const float* __restrict__ gA,
                             ushort_t* __restrict__ Bb) {
    int i = blockIdx.x * blockDim.x + threadIdx.x;        // group of 4 floats
    int h = i >> 8;                                       // (i*4) / D_DIM
    float g = gA[h];
    float4 f = ((const float4*)B)[i];
    uint_t lo = (uint_t)f2bf(f.x * g) | ((uint_t)f2bf(f.y * g) << 16);
    uint_t hi = (uint_t)f2bf(f.z * g) | ((uint_t)f2bf(f.w * g) << 16);
    ((uint2*)Bb)[i] = make_uint2(lo, hi);
}

// ---------------------------------------------------------------------------
// bf16 WMMA GEMM:  V[t][h] = sum_d Xb[t][d] * Bb[h][d]   (NT, both K-contiguous)
// Block tile 128(M) x 256(N) x 32(K); 8 waves as 2(M) x 4(N), each wave 64x64
// = 4x4 wmma tiles. Global -> LDS staging via CDNA5 async loads (ASYNCcnt)
// into a 3-stage ring: per K-step one barrier and a PARTIAL s_wait_asynccnt 6
// (one full stage of 6 fills stays in flight across the barrier).
// blockIdx.z selects (B1b->V1) or (B2b->V2).
// ---------------------------------------------------------------------------
#define BM 128
#define BN 256
#define BK 32
#define LDSW 40                 // padded row stride in shorts (32 data + 8 pad)
#define A_SH (BM * LDSW)        // 5120 shorts
#define B_SH (BN * LDSW)        // 10240 shorts
#define STAGE_SH (A_SH + B_SH)  // 15360 shorts per stage
#define NKSTEP (D_DIM / BK)     // 32

union FragU { v16bf v; v8bf h[2]; };

__device__ __forceinline__ void async_cp16(const ushort_t* lds_p, const ushort_t* g) {
    uint_t lds_addr = (uint_t)(size_t)lds_p;   // LDS aperture: flat addr[31:0]
    asm volatile("global_load_async_to_lds_b128 %0, %1, off"
                 :: "v"(lds_addr), "v"(g) : "memory");
}

__device__ __forceinline__ void wait_async0() {
    asm volatile("s_wait_asynccnt 0" ::: "memory");
}
__device__ __forceinline__ void wait_async6() {
    asm volatile("s_wait_asynccnt 6" ::: "memory");
}

__device__ __forceinline__ void stage_load(const ushort_t* __restrict__ Xb,
                                           const ushort_t* __restrict__ Bb,
                                           ushort_t* stage,
                                           int srow, int scolS,
                                           int bm, int bn, int k0) {
    // Each of 256 threads moves 6 x 16B segments: 2 for A (128x32), 4 for B (256x32)
#pragma unroll
    for (int j = 0; j < 2; j++) {
        int r = srow + j * 64;
        async_cp16(&stage[r * LDSW + scolS],
                   Xb + (size_t)(bm + r) * D_DIM + k0 + scolS);
    }
    ushort_t* stB = stage + A_SH;
#pragma unroll
    for (int j = 0; j < 4; j++) {
        int r = srow + j * 64;
        async_cp16(&stB[r * LDSW + scolS],
                   Bb + (size_t)(bn + r) * D_DIM + k0 + scolS);
    }
}

__global__ __launch_bounds__(256)
void gemm_kernel(const ushort_t* __restrict__ Xb,
                 const ushort_t* __restrict__ B1b,
                 const ushort_t* __restrict__ B2b,
                 float* __restrict__ V1, float* __restrict__ V2) {
    __shared__ ushort_t smem[3 * STAGE_SH];   // 90 KB, 3-stage ring

    const ushort_t* __restrict__ Bb = (blockIdx.z == 0) ? B1b : B2b;
    float* __restrict__ V           = (blockIdx.z == 0) ? V1  : V2;

    const int tid  = threadIdx.x;
    const int lane = tid & 31;
    const int wid  = tid >> 5;
    const int wm   = wid & 1;      // 0..1 -> 64 rows each
    const int wn   = wid >> 1;     // 0..3 -> 64 cols each
    const int bm   = blockIdx.y * BM;
    const int bn   = blockIdx.x * BN;

    v8f acc[4][4];
#pragma unroll
    for (int i = 0; i < 4; i++)
#pragma unroll
        for (int j = 0; j < 4; j++) acc[i][j] = (v8f)(0.0f);

    // async staging mapping: thread -> (row block, 16B column segment)
    const int srow  = tid >> 2;          // 0..63
    const int scolS = (tid & 3) * 8;     // short offset: 0,8,16,24

    // fragment addressing (ISA 7.12.2 bf16 A/B layout)
    const int rlo = lane & 15;           // row within 16
    const int kh  = (lane >> 4) * 8;     // lanes 0-15: K 0..7/16..23; 16-31: K 8..15/24..31

    ushort_t* st0 = smem;                 // current compute stage
    ushort_t* st1 = smem + STAGE_SH;      // next (in flight)
    ushort_t* st2 = smem + 2 * STAGE_SH;  // fill target

    stage_load(Xb, Bb, st0, srow, scolS, bm, bn, 0);
    stage_load(Xb, Bb, st1, srow, scolS, bm, bn, BK);

    for (int i = 0; i < NKSTEP; i++) {
        if (i + 1 < NKSTEP) wait_async6();   // oldest 6 (stage i) complete
        else                wait_async0();   // last stage: drain
        __syncthreads();                     // stage i visible to all waves;
                                             // all reads of stage (i-1) done
        if (i + 2 < NKSTEP)
            stage_load(Xb, Bb, st2, srow, scolS, bm, bn, (i + 2) * BK);

        const ushort_t* cA = st0;
        const ushort_t* cB = st0 + A_SH;

        FragU afr[4], bfr[4];
#pragma unroll
        for (int tm = 0; tm < 4; tm++) {
            int r = wm * 64 + tm * 16 + rlo;
            afr[tm].h[0] = *(const v8bf*)&cA[r * LDSW + kh];
            afr[tm].h[1] = *(const v8bf*)&cA[r * LDSW + kh + 16];
        }
#pragma unroll
        for (int tn = 0; tn < 4; tn++) {
            int r = wn * 64 + tn * 16 + rlo;
            bfr[tn].h[0] = *(const v8bf*)&cB[r * LDSW + kh];
            bfr[tn].h[1] = *(const v8bf*)&cB[r * LDSW + kh + 16];
        }
#pragma unroll
        for (int tm = 0; tm < 4; tm++)
#pragma unroll
            for (int tn = 0; tn < 4; tn++)
                acc[tm][tn] = __builtin_amdgcn_wmma_f32_16x16x32_bf16(
                    false, afr[tm].v, false, bfr[tn].v,
                    (short)0, acc[tm][tn], false, false);

        ushort_t* t = st0; st0 = st1; st1 = st2; st2 = t;   // rotate ring
    }

    // Epilogue: C layout -> lane = N + 16*(M>=8), vgpr r -> M = r (+8 hi lanes)
    const int nloc = lane & 15;
    const int mo   = (lane >> 4) * 8;
#pragma unroll
    for (int tm = 0; tm < 4; tm++)
#pragma unroll
        for (int tn = 0; tn < 4; tn++) {
            int gmb = bm + wm * 64 + tm * 16 + mo;
            int gn  = bn + wn * 64 + tn * 16 + nloc;
#pragma unroll
            for (int r = 0; r < 8; r++)
                V[(size_t)(gmb + r) * H_DIM + gn] = acc[tm][tn][r];
        }
}

// ---------------------------------------------------------------------------
// Scan phase E: per (chunk, channel) local scan from zero; record end states.
// ---------------------------------------------------------------------------
__global__ void ends_kernel(const float* __restrict__ V1, const float* __restrict__ V2,
                            const float* __restrict__ aA, const float* __restrict__ bA,
                            float* __restrict__ e1, float* __restrict__ e2) {
    int h = blockIdx.x * blockDim.x + threadIdx.x;
    int c = blockIdx.y;
    float a = aA[h], b = bA[h];
    float h1 = 0.0f, h2 = 0.0f;
    const float* p1 = V1 + (size_t)c * CLEN * H_DIM + h;
    const float* p2 = V2 + (size_t)c * CLEN * H_DIM + h;
#pragma unroll 4
    for (int t = 0; t < CLEN; t++) {
        float v1 = p1[(size_t)t * H_DIM];
        float v2 = p2[(size_t)t * H_DIM];
        float n1 = fmaf(a, h1, fmaf(-b, h2, v1));
        float n2 = fmaf(a, h2, fmaf( b, h1, v2));
        h1 = n1; h2 = n2;
    }
    e1[c * H_DIM + h] = h1;
    e2[c * H_DIM + h] = h2;
}

// ---------------------------------------------------------------------------
// Scan phase C: per channel, prefix over chunk ends using R^CLEN (7 squarings).
// ---------------------------------------------------------------------------
__global__ void carry_kernel(const float* __restrict__ aA, const float* __restrict__ bA,
                             const float* __restrict__ e1, const float* __restrict__ e2,
                             float* __restrict__ c1, float* __restrict__ c2) {
    int h = blockIdx.x * blockDim.x + threadIdx.x;
    float ar = aA[h], br = bA[h];
#pragma unroll
    for (int i = 0; i < 7; i++) {           // (a+ib)^(2^7) = R^CLEN
        float na = ar * ar - br * br;
        float nb = 2.0f * ar * br;
        ar = na; br = nb;
    }
    float p1 = 0.0f, p2 = 0.0f;
    for (int c = 0; c < CHUNKS; c++) {
        c1[c * H_DIM + h] = p1;
        c2[c * H_DIM + h] = p2;
        float v1 = e1[c * H_DIM + h];
        float v2 = e2[c * H_DIM + h];
        float n1 = fmaf(ar, p1, fmaf(-br, p2, v1));
        float n2 = fmaf(ar, p2, fmaf( br, p1, v2));
        p1 = n1; p2 = n2;
    }
}

// ---------------------------------------------------------------------------
// Scan phase F: re-scan each chunk seeded with its true carry; write output.
// ---------------------------------------------------------------------------
__global__ void final_kernel(const float* __restrict__ V1, const float* __restrict__ V2,
                             const float* __restrict__ aA, const float* __restrict__ bA,
                             const float* __restrict__ c1, const float* __restrict__ c2,
                             float* __restrict__ out) {
    int h = blockIdx.x * blockDim.x + threadIdx.x;
    int c = blockIdx.y;
    float a = aA[h], b = bA[h];
    float h1 = c1[c * H_DIM + h];
    float h2 = c2[c * H_DIM + h];
    const float* p1 = V1 + (size_t)c * CLEN * H_DIM + h;
    const float* p2 = V2 + (size_t)c * CLEN * H_DIM + h;
    float* po = out + (size_t)c * CLEN * (2 * H_DIM) + h;
#pragma unroll 2
    for (int t = 0; t < CLEN; t++) {
        float v1 = p1[(size_t)t * H_DIM];
        float v2 = p2[(size_t)t * H_DIM];
        float n1 = fmaf(a, h1, fmaf(-b, h2, v1));
        float n2 = fmaf(a, h2, fmaf( b, h1, v2));
        h1 = n1; h2 = n2;
        po[(size_t)t * 2 * H_DIM]         = h1;
        po[(size_t)t * 2 * H_DIM + H_DIM] = h2;
    }
}

// ---------------------------------------------------------------------------
extern "C" void kernel_launch(void* const* d_in, const int* in_sizes, int n_in,
                              void* d_out, int out_size, void* d_ws, size_t ws_size,
                              hipStream_t stream) {
    const float* X   = (const float*)d_in[0];
    const float* lam = (const float*)d_in[1];
    const float* th  = (const float*)d_in[2];
    const float* B1  = (const float*)d_in[3];
    const float* B2  = (const float*)d_in[4];
    float* out = (float*)d_out;

    char* ws = (char*)d_ws;
    size_t off = 0;
    auto alloc = [&](size_t bytes) -> char* {
        char* p = ws + off;
        off += (bytes + 255) & ~(size_t)255;
        return p;
    };
    ushort_t* Xb  = (ushort_t*)alloc((size_t)T_DIM * D_DIM * 2);
    ushort_t* B1b = (ushort_t*)alloc((size_t)H_DIM * D_DIM * 2);
    ushort_t* B2b = (ushort_t*)alloc((size_t)H_DIM * D_DIM * 2);
    float* aA = (float*)alloc(H_DIM * 4);
    float* bA = (float*)alloc(H_DIM * 4);
    float* gA = (float*)alloc(H_DIM * 4);
    float* V1 = (float*)alloc((size_t)T_DIM * H_DIM * 4);
    float* V2 = (float*)alloc((size_t)T_DIM * H_DIM * 4);
    float* e1 = (float*)alloc((size_t)CHUNKS * H_DIM * 4);
    float* e2 = (float*)alloc((size_t)CHUNKS * H_DIM * 4);
    float* c1 = (float*)alloc((size_t)CHUNKS * H_DIM * 4);
    float* c2 = (float*)alloc((size_t)CHUNKS * H_DIM * 4);
    (void)ws_size; (void)in_sizes; (void)n_in; (void)out_size;

    param_kernel<<<H_DIM / 256, 256, 0, stream>>>(lam, th, aA, bA, gA);
    cvt_x_kernel<<<(T_DIM * D_DIM / 4) / 256, 256, 0, stream>>>(X, Xb);
    cvt_b_kernel<<<(H_DIM * D_DIM / 4) / 256, 256, 0, stream>>>(B1, gA, B1b);
    cvt_b_kernel<<<(H_DIM * D_DIM / 4) / 256, 256, 0, stream>>>(B2, gA, B2b);

    gemm_kernel<<<dim3(H_DIM / BN, T_DIM / BM, 2), 256, 0, stream>>>(Xb, B1b, B2b, V1, V2);

    ends_kernel<<<dim3(H_DIM / 256, CHUNKS), 256, 0, stream>>>(V1, V2, aA, bA, e1, e2);
    carry_kernel<<<H_DIM / 256, 256, 0, stream>>>(aA, bA, e1, e2, c1, c2);
    final_kernel<<<dim3(H_DIM / 256, CHUNKS), 256, 0, stream>>>(V1, V2, aA, bA, c1, c2, out);
}
